// MultiheadAttPoolLayer_91345364451424
// MI455X (gfx1250) — matrix operand order repaired and checked by reading, not compiled, not run on hardware
//
#include <hip/hip_runtime.h>
#include <hip/hip_bf16.h>

// ---------------------------------------------------------------------------
// MultiheadAttPoolLayer on MI455X (gfx1250, wave32, WMMA + async-to-LDS).
//
//   qs   = q @ Wq^T + bq                          (128 x 1024)
//   qhat[b,h,:] = qs[b,h] @ Wk_h                  (bk cancels in softmax)
//   logits[b,h,l] = qhat[b,h,:] . k[b,l,:] / sqrt(128)
//   attn = softmax_l(logits)        -> written to d_out attn region (h,b,l)
//   pooled[b,h,:] = sum_l attn[b,h,l] * k[b,l,:]  (linearity: project after)
//   out[b,h,:] = pooled[b,h,:] @ Wv_h^T + bv_h    (attn sums to 1 -> +bv)
//
// Hot kernels (k_scores, k_pool) carry all k-matrix HBM traffic: they use
// GLOBAL_LOAD_ASYNC_TO_LDS_B128 with double-buffered LDS tiles so the next
// tile's HBM latency overlaps the current tile's WMMA math (ASYNCcnt +
// s_wait_asynccnt). Falls back to the sync global->VGPR->cvt->LDS path if the
// async builtin is unavailable in this toolchain.
// ---------------------------------------------------------------------------

typedef __attribute__((ext_vector_type(16))) _Float16 v16h;
typedef __attribute__((ext_vector_type(8)))  float    v8f;
typedef __attribute__((ext_vector_type(2)))  _Float16 h2;
typedef __attribute__((ext_vector_type(4)))  _Float16 h4;
typedef __attribute__((ext_vector_type(4)))  int      v4i;

#if __has_builtin(__builtin_amdgcn_global_load_async_to_lds_b128)
#define HAVE_ASYNC 1
#endif

#ifdef HAVE_ASYNC
// Builtin signature: void(v4i addrspace(1)*, v4i addrspace(3)*, Imm off, Imm cpol)
typedef __attribute__((address_space(1))) v4i gv4i;
typedef __attribute__((address_space(3))) v4i lv4i;
__device__ __forceinline__ void async_b128(const float* g, float* l) {
  __builtin_amdgcn_global_load_async_to_lds_b128((gv4i*)g, (lv4i*)l, 0, 0);
}
#if __has_builtin(__builtin_amdgcn_s_wait_asynccnt)
#define WAIT_ASYNC(n) __builtin_amdgcn_s_wait_asynccnt((short)(n))
#else
#define WAIT_ASYNC(n) asm volatile("s_wait_asynccnt %0" ::"i"(n) : "memory")
#endif
#endif

// A-matrix 16-bit 16x32 K index for (lane, element e): lanes 0-15 K=0..15, lanes 16-31 K=16..31
__device__ __forceinline__ int kmapA(int lane, int e) {
  return (lane & 16) | e;
}
// B-matrix 16-bit 32x16: lanes 0-15 hold K=0..15, lanes 16-31 hold K=16..31
__device__ __forceinline__ int kmapB(int lane, int e) {
  return (lane & 16) | e;
}

__device__ __forceinline__ v8f wmma16(v16h a, v16h b, v8f c) {
  return __builtin_amdgcn_wmma_f32_16x16x32_f16(false, a, false, b,
                                                (short)0, c, false, false);
}

// A fragment from LDS row-major [m][k] f16
__device__ __forceinline__ v16h fragA(const _Float16* rowBase, int lane) {
  v16h a;
#pragma unroll
  for (int p = 0; p < 8; ++p) {
    h2 v = *(const h2*)(rowBase + kmapA(lane, 2 * p));
    a[2 * p] = v[0]; a[2 * p + 1] = v[1];
  }
  return a;
}
// B fragment from LDS [n][k] f16 (32 contiguous k per n row)
__device__ __forceinline__ v16h fragB_nk(const _Float16* rowBase, int lane) {
  v16h b;
#pragma unroll
  for (int p = 0; p < 8; ++p) {
    h2 v = *(const h2*)(rowBase + (lane & 16) + 2 * p);
    b[2 * p] = v[0]; b[2 * p + 1] = v[1];
  }
  return b;
}
// B fragment from LDS [n][k] fp32 (async-staged raw rows), cvt at use
__device__ __forceinline__ v16h fragB_nk32(const float* rowBase, int lane) {
  v16h b;
  const float* p = rowBase + (lane & 16);
#pragma unroll
  for (int e = 0; e < 16; ++e) b[e] = (_Float16)p[e];
  return b;
}
// B fragment from LDS [k][n] f16 with row stride strideN
__device__ __forceinline__ v16h fragB_kn(const _Float16* colBase, int lane,
                                         int strideN) {
  v16h b;
#pragma unroll
  for (int e = 0; e < 16; ++e) b[e] = colBase[kmapB(lane, e) * strideN];
  return b;
}
// B fragment from LDS [k][n] fp32 with row stride strideN
__device__ __forceinline__ v16h fragB_kn32(const float* colBase, int lane,
                                           int strideN) {
  v16h b;
#pragma unroll
  for (int e = 0; e < 16; ++e)
    b[e] = (_Float16)colBase[((lane & 16) | e) * strideN];
  return b;
}

__device__ __forceinline__ void cvtStore4(_Float16* dst, float4 v) {
  h4 h;
  h[0] = (_Float16)v.x; h[1] = (_Float16)v.y;
  h[2] = (_Float16)v.z; h[3] = (_Float16)v.w;
  *(h4*)dst = h;
}

// ---------------------------------------------------------------------------
// K1: qs = q @ Wq^T + bq.  grid (mtile=8, ntile=8), M=16(b), N=128, K=1024
// ---------------------------------------------------------------------------
__global__ __launch_bounds__(256) void k_qs(const float* __restrict__ q,
                                            const float* __restrict__ Wq,
                                            const float* __restrict__ bq,
                                            float* __restrict__ qs) {
  __shared__ alignas(16) _Float16 As[16 * 32];
  __shared__ alignas(16) _Float16 Bs[128 * 32];   // [n][k]
  const int t = threadIdx.x, lane = t & 31, wave = t >> 5;
  const int m0 = blockIdx.x * 16, n0 = blockIdx.y * 128;
  v8f acc = {};
  for (int k0 = 0; k0 < 1024; k0 += 32) {
    __syncthreads();
    if (t < 64) {                         // A tile: 16x32
      int m = t >> 2, c = (t & 3) * 8;
      const float* src = q + (size_t)(m0 + m) * 1024 + k0 + c;
      cvtStore4(&As[m * 32 + c], *(const float4*)src);
      cvtStore4(&As[m * 32 + c + 4], *(const float4*)(src + 4));
    }
    {                                     // B tile: Wq rows n0+n, 32 k each
      int n = t >> 1, c = (t & 1) * 16;
      const float* src = Wq + (size_t)(n0 + n) * 1024 + k0 + c;
#pragma unroll
      for (int i = 0; i < 16; i += 4)
        cvtStore4(&Bs[n * 32 + c + i], *(const float4*)(src + i));
    }
    __syncthreads();
    v16h a = fragA(&As[(lane & 15) * 32], lane);
    v16h b = fragB_nk(&Bs[(wave * 16 + (lane & 15)) * 32], lane);
    acc = wmma16(a, b, acc);
  }
  const int n = n0 + wave * 16 + (lane & 15);
  const int mh = (lane & 16) ? 8 : 0;
  const float bias = bq[n];
#pragma unroll
  for (int i = 0; i < 8; ++i)
    qs[(size_t)(m0 + mh + i) * 1024 + n] = acc[i] + bias;
}

// ---------------------------------------------------------------------------
// K2: qhat[b,h,:] = qs[b, h*128:(h+1)*128] @ Wk_h.
// grid (h=8, mtile=8, ntile=8), M=16(b), N=128(dk), K=128(j)
// ---------------------------------------------------------------------------
__global__ __launch_bounds__(256) void k_qhat(const float* __restrict__ qs,
                                              const float* __restrict__ Wk,
                                              float* __restrict__ qhat) {
  __shared__ alignas(16) _Float16 As[16 * 32];
  __shared__ alignas(16) _Float16 Bs[32 * 128];   // [k][n]
  const int t = threadIdx.x, lane = t & 31, wave = t >> 5;
  const int h = blockIdx.x, m0 = blockIdx.y * 16, n0 = blockIdx.z * 128;
  v8f acc = {};
  for (int k0 = 0; k0 < 128; k0 += 32) {
    __syncthreads();
    if (t < 64) {
      int m = t >> 2, c = (t & 3) * 8;
      const float* src = qs + (size_t)(m0 + m) * 1024 + h * 128 + k0 + c;
      cvtStore4(&As[m * 32 + c], *(const float4*)src);
      cvtStore4(&As[m * 32 + c + 4], *(const float4*)(src + 4));
    }
    {                                     // Wk row j contiguous over n
      int r = t >> 3, c = (t & 7) * 16;
      const float* src = Wk + (size_t)(h * 128 + k0 + r) * 1024 + n0 + c;
#pragma unroll
      for (int i = 0; i < 16; i += 4)
        cvtStore4(&Bs[r * 128 + c + i], *(const float4*)(src + i));
    }
    __syncthreads();
    v16h a = fragA(&As[(lane & 15) * 32], lane);
    v16h b = fragB_kn(&Bs[wave * 16 + (lane & 15)], lane, 128);
    acc = wmma16(a, b, acc);
  }
  const int n = n0 + wave * 16 + (lane & 15);
  const int mh = (lane & 16) ? 8 : 0;
#pragma unroll
  for (int i = 0; i < 8; ++i)
    qhat[((size_t)(m0 + mh + i) * 8 + h) * 1024 + n] = acc[i];
}

// ---------------------------------------------------------------------------
// K3 (hot): logits[b,h,l] = qhat[b,h,:] . k[b,l,:] * (1/sqrt(128)).
// grid (b=128, ltile=8), M=16 (8 heads + 8 zero pad), N=256(l), K=1024(d)
// Async double-buffered k staging: HBM latency overlaps WMMA.
// ---------------------------------------------------------------------------
__global__ __launch_bounds__(256) void k_scores(const float* __restrict__ kin,
                                                const float* __restrict__ qhat,
                                                float* __restrict__ logits) {
  __shared__ alignas(16) _Float16 As[16 * 1024];  // full qhat block, 32 KB
#ifdef HAVE_ASYNC
  __shared__ alignas(16) float Bs32[2][256 * 32]; // [n=l][k=d] fp32, 2x32 KB
#else
  __shared__ alignas(16) _Float16 Bs[256 * 32];   // [n=l][k=d] f16, 16 KB
#endif
  const int t = threadIdx.x, lane = t & 31, wave = t >> 5;
  const int b = blockIdx.x, l0 = blockIdx.y * 256;
  {                                       // stage A once; rows 8..15 = 0
    int m = t >> 4, c = (t & 15) * 64;
    if (m < 8) {
      const float* src = qhat + ((size_t)b * 8 + m) * 1024 + c;
#pragma unroll
      for (int i = 0; i < 64; i += 4)
        cvtStore4(&As[m * 1024 + c + i], *(const float4*)(src + i));
    } else {
      h4 z = {};
#pragma unroll
      for (int i = 0; i < 64; i += 4) *(h4*)&As[m * 1024 + c + i] = z;
    }
  }
  v8f acc0 = {}, acc1 = {};
  const int nb = wave * 32;
  const float* kb = kin + ((size_t)b * 2048 + l0) * 1024;

#ifdef HAVE_ASYNC
  {                                       // prologue: issue tile 0
    const float* src = kb + (size_t)t * 1024;
    float* dst = &Bs32[0][t * 32];
#pragma unroll
    for (int i = 0; i < 32; i += 4) async_b128(src + i, dst + i);
  }
  for (int step = 0; step < 32; ++step) {
    const int k0 = step * 32;
    if (step + 1 < 32) {                  // issue next tile, then drain cur
      const float* src = kb + (size_t)t * 1024 + k0 + 32;
      float* dst = &Bs32[(step + 1) & 1][t * 32];
#pragma unroll
      for (int i = 0; i < 32; i += 4) async_b128(src + i, dst + i);
      WAIT_ASYNC(8);
    } else {
      WAIT_ASYNC(0);
    }
    __syncthreads();                      // tile(step) visible to all waves
    const float* Bcur = Bs32[step & 1];
    v16h a  = fragA(&As[(lane & 15) * 1024 + k0], lane);
    v16h b0 = fragB_nk32(&Bcur[(nb + (lane & 15)) * 32], lane);
    v16h b1 = fragB_nk32(&Bcur[(nb + 16 + (lane & 15)) * 32], lane);
    acc0 = wmma16(a, b0, acc0);
    acc1 = wmma16(a, b1, acc1);
    __syncthreads();                      // all reads done before overwrite
  }
#else
  for (int k0 = 0; k0 < 1024; k0 += 32) {
    __syncthreads();
    {                                     // thread t stages key row l0+t
      const float* src = kb + (size_t)t * 1024 + k0;
      _Float16* dst = &Bs[t * 32];
#pragma unroll
      for (int i = 0; i < 32; i += 4)
        cvtStore4(dst + i, *(const float4*)(src + i));
    }
    __syncthreads();
    v16h a  = fragA(&As[(lane & 15) * 1024 + k0], lane);
    v16h b0 = fragB_nk(&Bs[(nb + (lane & 15)) * 32], lane);
    v16h b1 = fragB_nk(&Bs[(nb + 16 + (lane & 15)) * 32], lane);
    acc0 = wmma16(a, b0, acc0);
    acc1 = wmma16(a, b1, acc1);
  }
#endif

  const float scale = 0.088388347648318447f;  // 1/sqrt(128)
  if (lane < 16) {                        // D rows 0..7 live in lanes 0..15
    size_t base = (size_t)b * 8 * 2048 + l0 + nb + lane;
#pragma unroll
    for (int i = 0; i < 8; ++i) {
      logits[base + (size_t)i * 2048]      = acc0[i] * scale;
      logits[base + (size_t)i * 2048 + 16] = acc1[i] * scale;
    }
  }
}

// ---------------------------------------------------------------------------
// K4: softmax over l per (b,h); writes attn directly in (h,b,l) output layout.
// ---------------------------------------------------------------------------
__global__ __launch_bounds__(256) void k_softmax(const float* __restrict__ logits,
                                                 float* __restrict__ attn) {
  const int r = blockIdx.x, h = r >> 7, b = r & 127, t = threadIdx.x;
  __shared__ float red[256];
  const float* src = logits + ((size_t)b * 8 + h) * 2048;
  float v[8];
  float mx = -3.0e38f;
#pragma unroll
  for (int i = 0; i < 8; ++i) { v[i] = src[t + i * 256]; mx = fmaxf(mx, v[i]); }
  red[t] = mx; __syncthreads();
  for (int s = 128; s > 0; s >>= 1) {
    if (t < s) red[t] = fmaxf(red[t], red[t + s]);
    __syncthreads();
  }
  mx = red[0]; __syncthreads();
  float sum = 0.f;
#pragma unroll
  for (int i = 0; i < 8; ++i) { v[i] = __expf(v[i] - mx); sum += v[i]; }
  red[t] = sum; __syncthreads();
  for (int s = 128; s > 0; s >>= 1) {
    if (t < s) red[t] += red[t + s];
    __syncthreads();
  }
  const float inv = 1.0f / red[0];
  float* dst = attn + (size_t)r * 2048;
#pragma unroll
  for (int i = 0; i < 8; ++i) dst[t + i * 256] = v[i] * inv;
}

// ---------------------------------------------------------------------------
// K5: pooled[b,h,:] = attn[b,h,:] @ k[b].
// grid (b=128, dtile=4), M=16 (8 heads), N=256(d), K=2048(l)
// Async double-buffered k staging as in K3.
// ---------------------------------------------------------------------------
__global__ __launch_bounds__(256) void k_pool(const float* __restrict__ kin,
                                              const float* __restrict__ attn,
                                              float* __restrict__ pooled) {
#ifdef HAVE_ASYNC
  __shared__ alignas(16) float Bs32[2][32 * 256]; // [k=l][n=d] fp32, 2x32 KB
#else
  __shared__ alignas(16) _Float16 Bs[32 * 256];   // [k=l][n=d] f16, 16 KB
#endif
  const int t = threadIdx.x, lane = t & 31, wave = t >> 5;
  const int b = blockIdx.x, d0 = blockIdx.y * 256;
  const int m = lane & 15;
  v8f acc0 = {}, acc1 = {};
  const int nb = wave * 32;
  const int sr = t >> 3, sc = (t & 7) * 32;       // staging row/col for thread
  const float* kbase = kin + ((size_t)b * 2048 + sr) * 1024 + d0 + sc;

#ifdef HAVE_ASYNC
  {                                       // prologue: issue tile 0
    float* dst = &Bs32[0][sr * 256 + sc];
#pragma unroll
    for (int i = 0; i < 32; i += 4) async_b128(kbase + i, dst + i);
  }
  for (int step = 0; step < 64; ++step) {
    const int l0 = step * 32;
    if (step + 1 < 64) {
      const float* src = kbase + (size_t)(l0 + 32) * 1024;
      float* dst = &Bs32[(step + 1) & 1][sr * 256 + sc];
#pragma unroll
      for (int i = 0; i < 32; i += 4) async_b128(src + i, dst + i);
      WAIT_ASYNC(8);
    } else {
      WAIT_ASYNC(0);
    }
    __syncthreads();
    v16h a = {};
    if (m < 8) {                          // attn row for head m, this batch
      const float* ap = attn + ((size_t)m * 128 + b) * 2048 + l0;
#pragma unroll
      for (int p = 0; p < 8; ++p) {
        int K = kmapA(lane, 2 * p);
        float2 w = *(const float2*)(ap + K);
        a[2 * p] = (_Float16)w.x; a[2 * p + 1] = (_Float16)w.y;
      }
    }
    const float* Bcur = Bs32[step & 1];
    v16h b0 = fragB_kn32(&Bcur[nb + m], lane, 256);
    v16h b1 = fragB_kn32(&Bcur[nb + 16 + m], lane, 256);
    acc0 = wmma16(a, b0, acc0);
    acc1 = wmma16(a, b1, acc1);
    __syncthreads();
  }
#else
  for (int l0 = 0; l0 < 2048; l0 += 32) {
    __syncthreads();
    {
      const float* src = kbase + (size_t)l0 * 1024;
      _Float16* dst = &Bs[sr * 256 + sc];
#pragma unroll
      for (int i = 0; i < 32; i += 4)
        cvtStore4(dst + i, *(const float4*)(src + i));
    }
    __syncthreads();
    v16h a = {};
    if (m < 8) {
      const float* ap = attn + ((size_t)m * 128 + b) * 2048 + l0;
#pragma unroll
      for (int p = 0; p < 8; ++p) {
        int K = kmapA(lane, 2 * p);
        float2 w = *(const float2*)(ap + K);
        a[2 * p] = (_Float16)w.x; a[2 * p + 1] = (_Float16)w.y;
      }
    }
    v16h b0 = fragB_kn(&Bs[nb + m], lane, 256);
    v16h b1 = fragB_kn(&Bs[nb + 16 + m], lane, 256);
    acc0 = wmma16(a, b0, acc0);
    acc1 = wmma16(a, b1, acc1);
  }
#endif

  if (lane < 16) {
    size_t base = (size_t)b * 8 * 1024 + d0 + nb + lane;
#pragma unroll
    for (int i = 0; i < 8; ++i) {
      pooled[base + (size_t)i * 1024]      = acc0[i];
      pooled[base + (size_t)i * 1024 + 16] = acc1[i];
    }
  }
}

// ---------------------------------------------------------------------------
// K6: out[b, h*128+n] = pooled[b,h,:] @ Wv_h^T + bv.
// grid (h=8, mtile=8), M=16(b), N=128(dv), K=1024(d)
// ---------------------------------------------------------------------------
__global__ __launch_bounds__(256) void k_outproj(const float* __restrict__ pooled,
                                                 const float* __restrict__ Wv,
                                                 const float* __restrict__ bv,
                                                 float* __restrict__ out) {
  __shared__ alignas(16) _Float16 As[16 * 32];
  __shared__ alignas(16) _Float16 Bs[128 * 32];   // [n][k]
  const int t = threadIdx.x, lane = t & 31, wave = t >> 5;
  const int h = blockIdx.x, m0 = blockIdx.y * 16;
  v8f acc = {};
  for (int k0 = 0; k0 < 1024; k0 += 32) {
    __syncthreads();
    if (t < 64) {
      int m = t >> 2, c = (t & 3) * 8;
      const float* src = pooled + ((size_t)(m0 + m) * 8 + h) * 1024 + k0 + c;
      cvtStore4(&As[m * 32 + c], *(const float4*)src);
      cvtStore4(&As[m * 32 + c + 4], *(const float4*)(src + 4));
    }
    {
      int n = t >> 1, c = (t & 1) * 16;
      const float* src = Wv + (size_t)(h * 128 + n) * 1024 + k0 + c;
#pragma unroll
      for (int i = 0; i < 16; i += 4)
        cvtStore4(&Bs[n * 32 + c + i], *(const float4*)(src + i));
    }
    __syncthreads();
    v16h a = fragA(&As[(lane & 15) * 32], lane);
    v16h bf = fragB_nk(&Bs[(wave * 16 + (lane & 15)) * 32], lane);
    acc = wmma16(a, bf, acc);
  }
  const int n = wave * 16 + (lane & 15);
  const int mh = (lane & 16) ? 8 : 0;
  const float bias = bv[h * 128 + n];
#pragma unroll
  for (int i = 0; i < 8; ++i)
    out[(size_t)(m0 + mh + i) * 1024 + h * 128 + n] = acc[i] + bias;
}

// ---------------------------------------------------------------------------
extern "C" void kernel_launch(void* const* d_in, const int* in_sizes, int n_in,
                              void* d_out, int out_size, void* d_ws,
                              size_t ws_size, hipStream_t stream) {
  (void)in_sizes; (void)n_in; (void)out_size; (void)ws_size;
  const float* q  = (const float*)d_in[0];
  const float* k  = (const float*)d_in[1];
  const float* Wq = (const float*)d_in[2];
  const float* bq = (const float*)d_in[3];
  const float* Wk = (const float*)d_in[4];
  // d_in[5] = bk: cancels inside softmax (constant over l) -> unused
  const float* Wv = (const float*)d_in[6];
  const float* bv = (const float*)d_in[7];

  float* out  = (float*)d_out;             // (128, 1024)
  float* attn = out + 128 * 1024;          // (8*128, 2048) head-major

  float* ws     = (float*)d_ws;            // ~16.5 MB scratch used
  float* qs     = ws;                      // 128*1024
  float* qhat   = qs + 128 * 1024;         // 128*8*1024
  float* logits = qhat + 128 * 8 * 1024;   // 128*8*2048
  float* pooled = logits + 128 * 8 * 2048; // 128*8*1024

  k_qs     <<<dim3(8, 8),    256, 0, stream>>>(q, Wq, bq, qs);
  k_qhat   <<<dim3(8, 8, 8), 256, 0, stream>>>(qs, Wk, qhat);
  k_scores <<<dim3(128, 8),  256, 0, stream>>>(k, qhat, logits);
  k_softmax<<<dim3(1024),    256, 0, stream>>>(logits, attn);
  k_pool   <<<dim3(128, 4),  256, 0, stream>>>(k, attn, pooled);
  k_outproj<<<dim3(8, 8),    256, 0, stream>>>(pooled, Wv, bv, out);
}
